// MultiHeadAttention_7954279432866
// MI455X (gfx1250) — matrix operand order repaired and checked by reading, not compiled, not run on hardware
//
#include <hip/hip_runtime.h>
#include <cstdint>
#include <cstddef>

#ifndef USE_TDM
#define USE_TDM 1
#endif

// ---------------- problem constants (match reference) ----------------
constexpr int B_   = 2;
constexpr int S_   = 2048;
constexpr int DIM_ = 2048;
constexpr int H_   = 16;
constexpr int KVH_ = 4;
constexpr int HD_  = 128;          // DIM/H
constexpr int GQ_  = H_ / KVH_;    // 4
constexpr float QK_SCALE = 0.08838834764831845f;   // 1/sqrt(128)
constexpr float NEGBIG   = -1.0e9f;

// ---------------- vector types ----------------
typedef __bf16 bf16_t;
typedef __attribute__((ext_vector_type(16))) __bf16 v16bf;
typedef __attribute__((ext_vector_type(8)))  __bf16 v8bf;
typedef __attribute__((ext_vector_type(8)))  float  v8f;
typedef __attribute__((ext_vector_type(4)))  uint32_t u32x4;
typedef __attribute__((ext_vector_type(8)))  uint32_t u32x8;

union bf16pair { v16bf v; v8bf h[2]; };

__device__ __forceinline__ v8f wmma_bf16(const v16bf& a, const v16bf& b, const v8f& c) {
  return __builtin_amdgcn_wmma_f32_16x16x32_bf16(false, a, false, b, (short)0, c, false, false);
}

#if USE_TDM
// byte offset of a static __shared__ object inside the workgroup LDS allocation
__device__ __forceinline__ uint32_t lds_off_u32(const void* p) {
  return (uint32_t)(uintptr_t)(__attribute__((address_space(3))) const char*)p;
}

// 2D TDM load: tile (tileH rows x tileW bf16 elems) from row-major tensor with
// row stride strideElems, into LDS at lds_byte_off with +8-elem row padding.
__device__ __forceinline__ void tdm_load_2d(uint32_t lds_byte_off, const void* gsrc,
                                            uint32_t tileW, uint32_t tileH,
                                            uint32_t strideElems,
                                            uint32_t tensorW, uint32_t tensorH) {
  const uint64_t ga = (uint64_t)(uintptr_t)gsrc;
  u32x4 g0;
  g0[0] = 1u;                                   // count=1 user descriptor
  g0[1] = lds_byte_off;                         // lds_addr
  g0[2] = (uint32_t)ga;                         // global_addr[31:0]
  g0[3] = (uint32_t)(ga >> 32) | 0x80000000u;   // global_addr[56:32] | type=2
  const uint32_t rowDw = (tileW * 2u) >> 2;               // dwords per row
  const uint32_t padIv = 31u - (uint32_t)__builtin_clz(rowDw) - 1u;  // 2<<code==rowDw
  const uint32_t padDw = 3u;                               // 4 dwords = 8 bf16 pad
  u32x8 g1;
  g1[0] = (1u << 16)            // data_size = 2 bytes
        | (1u << 20)            // pad_enable
        | (padIv << 22) | (padDw << 25);
  g1[1] = (tensorW & 0xFFFFu) << 16;                       // tensor_dim0[15:0]
  g1[2] = (tensorW >> 16) | ((tensorH & 0xFFFFu) << 16);   // td0[31:16] | td1[15:0]
  g1[3] = (tensorH >> 16) | (tileW << 16);                 // td1[31:16] | tile_dim0
  g1[4] = tileH;                                           // tile_dim1 (tile_dim2=0)
  g1[5] = strideElems;                                     // tensor_dim0_stride[31:0]
  g1[6] = 0u;
  g1[7] = 0u;
  asm volatile("tensor_load_to_lds %0, %1" :: "s"(g0), "s"(g1) : "memory");
}
#endif

// ---------------- fp32 -> bf16 convert ----------------
__global__ void cvt_f32_bf16_kernel(const float* __restrict__ x,
                                    bf16_t* __restrict__ y, int n) {
  int i = blockIdx.x * blockDim.x + threadIdx.x;
  if (i < n) y[i] = (bf16_t)x[i];
}

// transpose+convert: W fp32 [K,N] -> Wt bf16 [N,K]
__global__ void wt_cvt_kernel(const float* __restrict__ x, bf16_t* __restrict__ y,
                              int K, int N) {
  int i = blockIdx.x * blockDim.x + threadIdx.x;
  if (i >= K * N) return;
  int k = i / N, n = i - k * N;
  y[(size_t)n * K + k] = (bf16_t)x[i];
}

// V fp32 [B,S,KVH,HD] -> Vt bf16 [B,KVH,HD,S]
__global__ void vt_cvt_kernel(const float* __restrict__ x, bf16_t* __restrict__ y) {
  int i = blockIdx.x * blockDim.x + threadIdx.x;
  if (i >= B_ * S_ * KVH_ * HD_) return;
  int d   = i & (HD_ - 1);
  int kvh = (i >> 7) & (KVH_ - 1);
  int s   = (i >> 9) & (S_ - 1);
  int b   = i >> 20;
  y[(((size_t)b * KVH_ + kvh) * HD_ + d) * S_ + s] = (bf16_t)x[i];
}

// ---------------- RoPE + scale + bf16 convert ----------------
__global__ void rope_cvt_kernel(const float* __restrict__ X,
                                const float* __restrict__ cosT,
                                const float* __restrict__ sinT,
                                bf16_t* __restrict__ Y,
                                int heads, float outScale, int total) {
  int i = blockIdx.x * blockDim.x + threadIdx.x;
  if (i >= total) return;
  int d = i & (HD_ - 1);
  int t = i / HD_;
  int s = (t / heads) & (S_ - 1);
  float x = X[i];
  float partner = (d < HD_ / 2) ? -X[i + HD_ / 2] : X[i - HD_ / 2];
  Y[i] = (bf16_t)((x * cosT[s * HD_ + d] + partner * sinT[s * HD_ + d]) * outScale);
}

// ---------------- bf16 WMMA GEMM:  C[M,N] = A[M,K] @ Wt[N,K]^T ----------------
// BM=64 BN=128 BK=64, 256 threads = 8 waves, wave computes 16 rows x 64 cols.
#if USE_TDM
#define GEMM_BUFS 2
#else
#define GEMM_BUFS 1
#endif
__global__ __launch_bounds__(256)
void gemm_bf16_kernel(const bf16_t* __restrict__ A, const bf16_t* __restrict__ Wt,
                      float* __restrict__ C, int M, int N, int K) {
  __shared__ __align__(16) bf16_t lA[GEMM_BUFS][64][72];    // rows x (64+8 pad)
  __shared__ __align__(16) bf16_t lB[GEMM_BUFS][128][72];   // cols x (64+8 pad)
  const int tid  = threadIdx.x;
  const int lane = tid & 31;
  const int wave = tid >> 5;
  const int l16  = lane & 15;
  const int lh   = lane >> 4;
  const int mblk = blockIdx.y * 64;
  const int nblk = blockIdx.x * 128;
  const int wm   = (wave & 3) * 16;
  const int wn   = (wave >> 2) * 64;
  const int koff = lh * 16;   // A-frag per-half K offset (contiguous, matches B)
  const int kb   = lh * 16;   // B-frag per-half K offset (contiguous, matches A)

  v8f acc[4] = {};
  int buf = 0;

#if USE_TDM
  if (wave == 0) {
    tdm_load_2d(lds_off_u32(&lA[0][0][0]), A + (size_t)mblk * K, 64, 64,
                (uint32_t)K, (uint32_t)K, (uint32_t)M);
    tdm_load_2d(lds_off_u32(&lB[0][0][0]), Wt + (size_t)nblk * K, 64, 128,
                (uint32_t)K, (uint32_t)K, (uint32_t)N);
  }
#endif

  for (int k0 = 0; k0 < K; k0 += 64) {
#if USE_TDM
    // own-wave TENSORcnt: no-op for waves that issued nothing -> no branch needed
    __builtin_amdgcn_s_wait_tensorcnt((short)0);
    __syncthreads();
    if (wave == 0 && (k0 + 64) < K) {   // prefetch next tiles via TDM (overlaps WMMAs)
      const int nb = buf ^ 1;
      tdm_load_2d(lds_off_u32(&lA[nb][0][0]), A + (size_t)mblk * K + (k0 + 64),
                  64, 64, (uint32_t)K, (uint32_t)K, (uint32_t)M);
      tdm_load_2d(lds_off_u32(&lB[nb][0][0]), Wt + (size_t)nblk * K + (k0 + 64),
                  64, 128, (uint32_t)K, (uint32_t)K, (uint32_t)N);
    }
#else
    {   // manual staging: contiguous row copies (Wt pre-transposed -> no b16 stores)
      const int ar = tid >> 2, ac = (tid & 3) * 16;   // A: 64x64, 32B/thread
      *(v8bf*)&lA[0][ar][ac]     = *(const v8bf*)(A + (size_t)(mblk + ar) * K + k0 + ac);
      *(v8bf*)&lA[0][ar][ac + 8] = *(const v8bf*)(A + (size_t)(mblk + ar) * K + k0 + ac + 8);
      const int br = tid >> 1, bc = (tid & 1) * 32;   // B: 128x64, 64B/thread
      #pragma unroll
      for (int q = 0; q < 4; ++q)
        *(v8bf*)&lB[0][br][bc + q * 8] =
            *(const v8bf*)(Wt + (size_t)(nblk + br) * K + k0 + bc + q * 8);
    }
    __syncthreads();
#endif

    // batch ALL fragment loads, then issue the 8 WMMAs back-to-back
    bf16pair ua[2], ub[2][4];
    #pragma unroll
    for (int c = 0; c < 2; ++c) {
      ua[c].h[0] = *(const v8bf*)&lA[buf][wm + l16][c * 32 + koff];
      ua[c].h[1] = *(const v8bf*)&lA[buf][wm + l16][c * 32 + koff + 8];
      #pragma unroll
      for (int j = 0; j < 4; ++j) {
        ub[c][j].h[0] = *(const v8bf*)&lB[buf][wn + j * 16 + l16][c * 32 + kb];
        ub[c][j].h[1] = *(const v8bf*)&lB[buf][wn + j * 16 + l16][c * 32 + kb + 8];
      }
    }
    #pragma unroll
    for (int c = 0; c < 2; ++c)
      #pragma unroll
      for (int j = 0; j < 4; ++j)
        acc[j] = wmma_bf16(ua[c].v, ub[c][j].v, acc[j]);
    __syncthreads();
#if USE_TDM
    buf ^= 1;
#endif
  }

  #pragma unroll
  for (int j = 0; j < 4; ++j) {
    const int col   = nblk + wn + j * 16 + l16;
    const int rbase = mblk + wm + lh * 8;
    #pragma unroll
    for (int r = 0; r < 8; ++r)
      C[(size_t)(rbase + r) * N + col] = acc[j][r];
  }
}

// ---------------- flash attention, 1 wave per (b, h, 16-row q tile) ----------------
// Qb: [B,S,H,HD] bf16 (pre-scaled 1/sqrt(HD)); Kb: [B,S,KVH,HD] bf16;
// Vt: [B,KVH,HD,S] bf16; alibi [B,S,S] f32; amask [B,1,1,S] f32; Of [B,S,H,HD] f32.
__global__ __launch_bounds__(32)
void attn_fa_kernel(const bf16_t* __restrict__ Qb, const bf16_t* __restrict__ Kb,
                    const bf16_t* __restrict__ Vt, const float* __restrict__ alibi,
                    const float* __restrict__ amask, float* __restrict__ Of) {
  __shared__ __align__(16) bf16_t lP[16][40];    // P tile [row][key]
  const int lane = threadIdx.x;
  const int l16  = lane & 15;
  const int lh   = lane >> 4;
  const int qt = blockIdx.x, h = blockIdx.y, b = blockIdx.z;
  const int kvh   = h / GQ_;
  const int qbase = qt * 16;
  const int koff  = lh * 16;
  const int kb    = lh * 16;

  v16bf qf[4];
  {
    const bf16_t* qrow = Qb + (((size_t)b * S_ + (qbase + l16)) * H_ + h) * HD_;
    #pragma unroll
    for (int c = 0; c < 4; ++c) {
      bf16pair u;
      u.h[0] = *(const v8bf*)(qrow + c * 32 + koff);
      u.h[1] = *(const v8bf*)(qrow + c * 32 + koff + 8);
      qf[c] = u.v;
    }
  }

  v8f o[8] = {};
  float mrow[8], lrow[8];
  #pragma unroll
  for (int r = 0; r < 8; ++r) { mrow[r] = -3.0e38f; lrow[r] = 0.0f; }

  const bf16_t* vbase = Vt + (size_t)(b * KVH_ + kvh) * HD_ * S_;
  const int kend = qbase + 16;
  for (int kt = 0; kt < kend; kt += 32) {
    // scores for keys kt..kt+31 (two 16x16 tiles)
    v8f s0 = {}, s1 = {};
    {
      const bf16_t* kr0 = Kb + (((size_t)b * S_ + (kt + l16)) * KVH_ + kvh) * HD_ + kb;
      const bf16_t* kr1 = kr0 + (size_t)16 * KVH_ * HD_;
      bf16pair u0[4], u1[4];
      #pragma unroll
      for (int c = 0; c < 4; ++c) {
        u0[c].h[0] = *(const v8bf*)(kr0 + c * 32);
        u0[c].h[1] = *(const v8bf*)(kr0 + c * 32 + 8);
        u1[c].h[0] = *(const v8bf*)(kr1 + c * 32);
        u1[c].h[1] = *(const v8bf*)(kr1 + c * 32 + 8);
      }
      #pragma unroll
      for (int c = 0; c < 4; ++c) {
        s0 = wmma_bf16(qf[c], u0[c].v, s0);
        s1 = wmma_bf16(qf[c], u1[c].v, s1);
      }
    }

    // bias + causal mask + online softmax (row stats across 16-lane halves)
    const int col0 = kt + l16;
    const int col1 = col0 + 16;
    const float am0 = amask[(size_t)b * S_ + col0];
    const float am1 = amask[(size_t)b * S_ + col1];
    #pragma unroll
    for (int r = 0; r < 8; ++r) {
      const int sq = qbase + r + lh * 8;
      const float* arow = alibi + ((size_t)b * S_ + sq) * S_;
      float v0 = s0[r] + arow[col0] + am0;
      float v1 = s1[r] + arow[col1] + am1;
      if (col0 > sq) v0 += NEGBIG;
      if (col1 > sq) v1 += NEGBIG;
      float tm = fmaxf(v0, v1);
      #pragma unroll
      for (int off = 1; off < 16; off <<= 1) tm = fmaxf(tm, __shfl_xor(tm, off, 16));
      const float nm   = fmaxf(mrow[r], tm);
      const float corr = __expf(mrow[r] - nm);
      const float p0 = __expf(v0 - nm);
      const float p1 = __expf(v1 - nm);
      float ts = p0 + p1;
      #pragma unroll
      for (int off = 1; off < 16; off <<= 1) ts += __shfl_xor(ts, off, 16);
      lrow[r] = lrow[r] * corr + ts;
      mrow[r] = nm;
      #pragma unroll
      for (int j = 0; j < 8; ++j) o[j][r] *= corr;
      lP[r + lh * 8][l16]      = (bf16_t)p0;
      lP[r + lh * 8][16 + l16] = (bf16_t)p1;
    }
    __syncthreads();

    // O += P(16x32) @ V(32x128); B-frags straight from pre-transposed Vt
    bf16pair up;
    up.h[0] = *(const v8bf*)&lP[l16][koff];
    up.h[1] = *(const v8bf*)&lP[l16][koff + 8];
    bf16pair uv[8];
    #pragma unroll
    for (int j = 0; j < 8; ++j) {
      const bf16_t* vrow = vbase + (size_t)(j * 16 + l16) * S_ + kt + kb;
      uv[j].h[0] = *(const v8bf*)(vrow);
      uv[j].h[1] = *(const v8bf*)(vrow + 8);
    }
    #pragma unroll
    for (int j = 0; j < 8; ++j) o[j] = wmma_bf16(up.v, uv[j].v, o[j]);
    __syncthreads();
  }

  #pragma unroll
  for (int r = 0; r < 8; ++r) {
    const int sq = qbase + r + lh * 8;
    const float inv = 1.0f / lrow[r];
    float* orow = Of + (((size_t)b * S_ + sq) * H_ + h) * HD_ + l16;
    #pragma unroll
    for (int j = 0; j < 8; ++j) orow[j * 16] = o[j][r] * inv;
  }
}

// ---------------- host-side orchestration ----------------
extern "C" void kernel_launch(void* const* d_in, const int* in_sizes, int n_in,
                              void* d_out, int out_size, void* d_ws, size_t ws_size,
                              hipStream_t stream) {
  (void)in_sizes; (void)n_in; (void)out_size; (void)ws_size;
  const float* hidden = (const float*)d_in[0];
  const float* pcos   = (const float*)d_in[1];
  const float* psin   = (const float*)d_in[2];
  const float* alibi  = (const float*)d_in[3];
  const float* amask  = (const float*)d_in[4];
  const float* wq     = (const float*)d_in[5];
  const float* wk     = (const float*)d_in[6];
  const float* wv     = (const float*)d_in[7];
  const float* wo     = (const float*)d_in[8];
  float* out = (float*)d_out;
  char*  ws  = (char*)d_ws;

  const int M = B_ * S_;                 // 4096
  bf16_t* Xb  = (bf16_t*)(ws + 0);                        // 16,777,216
  bf16_t* Wqb = (bf16_t*)(ws + 16777216);                 //  8,388,608 (transposed)
  bf16_t* Wkb = (bf16_t*)(ws + 25165824);                 //  2,097,152 (transposed)
  bf16_t* Wvb = (bf16_t*)(ws + 27262976);                 //  2,097,152 (transposed)
  bf16_t* Wob = (bf16_t*)(ws + 29360128);                 //  8,388,608 (transposed)
  float*  Qf  = (float*) (ws + 37748736);                 // 33,554,432 (reused as Of)
  float*  Kf  = (float*) (ws + 71303168);                 //  8,388,608 (reused as Ob)
  float*  Vf  = (float*) (ws + 79691776);                 //  8,388,608
  bf16_t* Qb  = (bf16_t*)(ws + 88080384);                 // 16,777,216
  bf16_t* Kb  = (bf16_t*)(ws + 104857600);                //  4,194,304
  bf16_t* Vt  = (bf16_t*)(ws + 109051904);                //  4,194,304
  float*  Of  = Qf;
  bf16_t* Ob  = (bf16_t*)Kf;

  auto cvt = [&](const float* src, bf16_t* dst, int n) {
    cvt_f32_bf16_kernel<<<(n + 255) / 256, 256, 0, stream>>>(src, dst, n);
  };
  auto wtc = [&](const float* src, bf16_t* dst, int K, int N) {
    wt_cvt_kernel<<<(K * N + 255) / 256, 256, 0, stream>>>(src, dst, K, N);
  };

  // 1) conversions (weights transposed to [N,K])
  cvt(hidden, Xb, M * DIM_);
  wtc(wq, Wqb, DIM_, DIM_);
  wtc(wk, Wkb, DIM_, KVH_ * HD_);
  wtc(wv, Wvb, DIM_, KVH_ * HD_);
  wtc(wo, Wob, DIM_, DIM_);

  // 2) projections (TDM-staged WMMA GEMMs)
  gemm_bf16_kernel<<<dim3(DIM_ / 128, M / 64), 256, 0, stream>>>(
      Xb, Wqb, Qf, M, DIM_, DIM_);
  gemm_bf16_kernel<<<dim3((KVH_ * HD_) / 128, M / 64), 256, 0, stream>>>(
      Xb, Wkb, Kf, M, KVH_ * HD_, DIM_);
  gemm_bf16_kernel<<<dim3((KVH_ * HD_) / 128, M / 64), 256, 0, stream>>>(
      Xb, Wvb, Vf, M, KVH_ * HD_, DIM_);

  // 3) RoPE (+fold 1/sqrt(HD) into Q) and layout conversions
  {
    int nq = M * DIM_;
    rope_cvt_kernel<<<(nq + 255) / 256, 256, 0, stream>>>(
        Qf, pcos, psin, Qb, H_, QK_SCALE, nq);
    int nk = M * KVH_ * HD_;
    rope_cvt_kernel<<<(nk + 255) / 256, 256, 0, stream>>>(
        Kf, pcos, psin, Kb, KVH_, 1.0f, nk);
    vt_cvt_kernel<<<(nk + 255) / 256, 256, 0, stream>>>(Vf, Vt);
  }

  // 4) flash attention (WMMA QK^T and PV)
  attn_fa_kernel<<<dim3(S_ / 16, H_, B_), 32, 0, stream>>>(
      Qb, Kb, Vt, alibi, amask, Of);

  // 5) output projection
  cvt(Of, Ob, M * DIM_);
  gemm_bf16_kernel<<<dim3(DIM_ / 128, M / 64), 256, 0, stream>>>(
      Ob, Wob, out, M, DIM_, DIM_);
}
